// AxialAttention_22806276342015
// MI455X (gfx1250) — matrix operand order repaired and checked by reading, not compiled, not run on hardware
//
#include <hip/hip_runtime.h>

// Problem constants (match reference)
#define BB   16
#define CC   512
#define C8C  64
#define HHH  64
#define WWW  64
#define HWSZ 4096
#define RELLEN 512

// LDS layout (halfword strides padded to keep 16B alignment + spread banks)
#define XPAD 544   // Xs row stride (halfs): 64 rows of [w][c]
#define QPAD 72    // Qs/Ks/Ps/Vs row stride (halfs)
#define SPAD 66    // Ss row stride (floats)

#define XS_OFF 0
#define QS_OFF (XS_OFF + WWW * XPAD * 2)          // 69632
#define KS_OFF (QS_OFF + WWW * QPAD * 2)          // +9216
#define PS_OFF (KS_OFF + WWW * QPAD * 2)
#define VS_OFF (PS_OFF + WWW * QPAD * 2)
#define SS_OFF (VS_OFF + 64  * QPAD * 2)
#define SMEM_BYTES (SS_OFF + WWW * SPAD * 4)      // 123392 bytes

typedef __attribute__((ext_vector_type(16))) __bf16        v16bf;
typedef __attribute__((ext_vector_type(8)))  float         v8f;
typedef __attribute__((ext_vector_type(4)))  unsigned int  u32x4;

#define VZERO {0.f, 0.f, 0.f, 0.f, 0.f, 0.f, 0.f, 0.f}

union FragU {
    v16bf bf;
    u32x4 q[2];
    unsigned short us[16];
};

static __device__ __forceinline__ unsigned short f2bf(float f) {
    unsigned u = __builtin_bit_cast(unsigned, f);
    u += 0x7fffu + ((u >> 16) & 1u);          // round-to-nearest-even
    return (unsigned short)(u >> 16);
}

// A fragment: 32 contiguous bytes (lane offset 16*hi pre-applied)
static __device__ __forceinline__ FragU ld_frag_a(const unsigned short* p) {
    FragU a;
    a.q[0] = *(const u32x4*)(p);
    a.q[1] = *(const u32x4*)(p + 8);
    return a;
}
// B fragment: 32 contiguous bytes (lane offset 16*hi pre-applied)
static __device__ __forceinline__ FragU ld_frag_b(const unsigned short* p) {
    FragU b;
    b.q[0] = *(const u32x4*)(p);
    b.q[1] = *(const u32x4*)(p + 8);
    return b;
}
static __device__ __forceinline__ v8f wmma_bf16(FragU a, FragU b, v8f c) {
    return __builtin_amdgcn_wmma_f32_16x16x32_bf16(false, a.bf, false, b.bf,
                                                   (short)0, c, false, false);
}

// ---------------------------------------------------------------------------
// Kernel 1: one-time f32 -> bf16 conversion of Wq, Wk, Wv into workspace.
// Layout in ws (ushorts): [Wq 64x512][Wk 64x512][Wv 512x512]
// ---------------------------------------------------------------------------
__global__ void convert_weights_kernel(const float* __restrict__ Wq,
                                       const float* __restrict__ Wk,
                                       const float* __restrict__ Wv,
                                       unsigned short* __restrict__ wsb) {
    int i = blockIdx.x * blockDim.x + threadIdx.x;
    const int nqk = C8C * CC;                  // 32768
    if (i < nqk)                  wsb[i] = f2bf(Wq[i]);
    else if (i < 2 * nqk)         wsb[i] = f2bf(Wk[i - nqk]);
    else if (i < 2 * nqk + CC*CC) wsb[i] = f2bf(Wv[i - 2 * nqk]);
}

// ---------------------------------------------------------------------------
// Kernel 2: fully fused axial attention. One workgroup per (b, h).
// Each wave runs ONE shared A-fragment stream against 2-4 independent
// B-fragment streams / accumulator chains (independent WMMA dep chains).
// ---------------------------------------------------------------------------
__global__ void __launch_bounds__(256)
axial_attn_kernel(const float* __restrict__ x,
                  const unsigned short* __restrict__ wqb,
                  const unsigned short* __restrict__ wkb,
                  const unsigned short* __restrict__ wvb,
                  const float* __restrict__ bq,
                  const float* __restrict__ bk,
                  const float* __restrict__ bv,
                  const float* __restrict__ rel_h,
                  const float* __restrict__ rel_w,
                  float* __restrict__ out) {
    extern __shared__ char smem[];
    unsigned short* Xs = (unsigned short*)(smem + XS_OFF);  // [w][c] bf16 (X^T)
    unsigned short* Qs = (unsigned short*)(smem + QS_OFF);  // [w][o] bf16 (Q^T)
    unsigned short* Ks = (unsigned short*)(smem + KS_OFF);  // [v][o] bf16 (K^T)
    unsigned short* Ps = (unsigned short*)(smem + PS_OFF);  // [w][v] bf16 softmax(S)
    unsigned short* Vs = (unsigned short*)(smem + VS_OFF);  // [d][v] bf16 (V block)
    float*          Ss = (float*)(smem + SS_OFF);           // [w][v] f32 scores

    const int tid  = threadIdx.x;
    const int lane = tid & 31;
    const int wvid = tid >> 5;       // wave id 0..7
    const int ln   = lane & 15;
    const int hi   = lane >> 4;      // lane group (0: lanes 0-15, 1: 16-31)

    const int bh = blockIdx.x;
    const int b  = bh >> 6;
    const int h  = bh & 63;

    const float* xb = x + (size_t)b * CC * HWSZ + (size_t)h * WWW;

    // ---- Phase 0: load x[b,:,h,:], convert to bf16, store transposed ----
    for (int idx = tid; idx < CC * (WWW / 4); idx += 256) {
        int c  = idx >> 4;
        int w4 = (idx & 15) << 2;
        const float4 v = *(const float4*)(xb + (size_t)c * HWSZ + w4);
        Xs[(w4 + 0) * XPAD + c] = f2bf(v.x);
        Xs[(w4 + 1) * XPAD + c] = f2bf(v.y);
        Xs[(w4 + 2) * XPAD + c] = f2bf(v.z);
        Xs[(w4 + 3) * XPAD + c] = f2bf(v.w);
    }
    __syncthreads();

    // ---- Phase 1: Q = Wq*X + bq + rel_h ; K = Wk*X + bk + rel_w ----
    // Waves 0-3: Q rows o0 = 16*wv.  Waves 4-7: K rows o0 = 16*(wv-4).
    // One A stream (weights), 4 B streams (all four 16-wide w tiles).
    {
        const int isK = wvid >> 2;
        const int o0  = (wvid & 3) << 4;
        const unsigned short* Wb   = isK ? wkb : wqb;
        const unsigned short* arow = Wb + (size_t)(o0 + ln) * CC + 16 * hi;
        const unsigned short* b0   = Xs + ( 0 + ln) * XPAD + 16 * hi;
        const unsigned short* b1   = Xs + (16 + ln) * XPAD + 16 * hi;
        const unsigned short* b2   = Xs + (32 + ln) * XPAD + 16 * hi;
        const unsigned short* b3   = Xs + (48 + ln) * XPAD + 16 * hi;
        v8f acc0 = VZERO, acc1 = VZERO, acc2 = VZERO, acc3 = VZERO;
        for (int k0 = 0; k0 < CC; k0 += 32) {
            FragU a = ld_frag_a(arow + k0);
            acc0 = wmma_bf16(a, ld_frag_b(b0 + k0), acc0);
            acc1 = wmma_bf16(a, ld_frag_b(b1 + k0), acc1);
            acc2 = wmma_bf16(a, ld_frag_b(b2 + k0), acc2);
            acc3 = wmma_bf16(a, ld_frag_b(b3 + k0), acc3);
        }
        v8f accs[4] = {acc0, acc1, acc2, acc3};
        #pragma unroll
        for (int j = 0; j < 4; ++j) {
            const int n0 = j << 4;
            FragU pk;
            if (!isK) {
                #pragma unroll
                for (int r = 0; r < 8; ++r) {
                    int o = o0 + 8 * hi + r;
                    pk.us[r] = f2bf(accs[j][r] + bq[o] + rel_h[o * RELLEN + h]);
                }
                *(u32x4*)(Qs + (n0 + ln) * QPAD + o0 + 8 * hi) = pk.q[0];
            } else {
                #pragma unroll
                for (int r = 0; r < 8; ++r) {
                    int o = o0 + 8 * hi + r;
                    pk.us[r] = f2bf(accs[j][r] + bk[o] + rel_w[o * RELLEN + (n0 + ln)]);
                }
                *(u32x4*)(Ks + (n0 + ln) * QPAD + o0 + 8 * hi) = pk.q[0];
            }
        }
    }
    __syncthreads();

    // ---- Phase 2: S[w,v] = sum_o Q^T[w][o] * K^T[v][o] ----
    // Wave -> (w0 = 16*(wv>>1), v0 = 32*(wv&1)): one A stream, 2 B streams.
    {
        const int w0 = (wvid >> 1) << 4;
        const int v0 = (wvid & 1) << 5;
        const unsigned short* arow = Qs + (w0 + ln) * QPAD + 16 * hi;
        const unsigned short* b0   = Ks + (v0 + ln) * QPAD + 16 * hi;
        const unsigned short* b1   = Ks + (v0 + 16 + ln) * QPAD + 16 * hi;
        v8f acc0 = VZERO, acc1 = VZERO;
        #pragma unroll
        for (int k0 = 0; k0 < C8C; k0 += 32) {
            FragU a = ld_frag_a(arow + k0);
            acc0 = wmma_bf16(a, ld_frag_b(b0 + k0), acc0);
            acc1 = wmma_bf16(a, ld_frag_b(b1 + k0), acc1);
        }
        #pragma unroll
        for (int r = 0; r < 8; ++r) {
            Ss[(w0 + 8 * hi + r) * SPAD + v0 + ln]      = acc0[r];
            Ss[(w0 + 8 * hi + r) * SPAD + v0 + 16 + ln] = acc1[r];
        }
    }
    __syncthreads();

    // ---- Phase 3: row softmax, P -> bf16 ----
    if (tid < WWW) {
        float row[WWW];
        float mx = -1e30f;
        for (int v = 0; v < WWW; ++v) {
            row[v] = Ss[tid * SPAD + v];
            mx = fmaxf(mx, row[v]);
        }
        float sum = 0.f;
        for (int v = 0; v < WWW; ++v) {
            float e = __expf(row[v] - mx);
            row[v] = e;
            sum += e;
        }
        float inv = 1.0f / sum;
        for (int v = 0; v < WWW; ++v)
            Ps[tid * QPAD + v] = f2bf(row[v] * inv);
    }
    __syncthreads();

    // ---- Phase 4: for each 64-channel d-block: V proj then AV ----
    const int m0 = (wvid >> 1) << 4;     // d tile within block
    const int n0 = (wvid & 1) << 5;      // v (4a) / w (4b) tile base
    for (int dblk = 0; dblk < 8; ++dblk) {
        const int dbase = dblk << 6;

        // 4a: V_blk[d][v] = Wv[dbase+d,:] * X[:,v] + bv
        {
            const unsigned short* arow =
                wvb + (size_t)(dbase + m0 + ln) * CC + 16 * hi;
            const unsigned short* b0 = Xs + (n0 + ln) * XPAD + 16 * hi;
            const unsigned short* b1 = Xs + (n0 + 16 + ln) * XPAD + 16 * hi;
            if (dblk < 7)  // warm next block's weight rows into cache
                __builtin_prefetch(wvb + (size_t)(dbase + 64 + m0 + ln) * CC, 0, 1);
            v8f acc0 = VZERO, acc1 = VZERO;
            for (int k0 = 0; k0 < CC; k0 += 32) {
                FragU a = ld_frag_a(arow + k0);
                acc0 = wmma_bf16(a, ld_frag_b(b0 + k0), acc0);
                acc1 = wmma_bf16(a, ld_frag_b(b1 + k0), acc1);
            }
            #pragma unroll
            for (int r = 0; r < 8; ++r) {
                int d = m0 + 8 * hi + r;
                float bias = bv[dbase + d];
                Vs[d * QPAD + n0 + ln]      = f2bf(acc0[r] + bias);
                Vs[d * QPAD + n0 + 16 + ln] = f2bf(acc1[r] + bias);
            }
        }
        __syncthreads();

        // 4b: Out[d,w] = sum_v V_blk[d][v] * P[w][v] ; store straight to HBM
        {
            const unsigned short* arow = Vs + (m0 + ln) * QPAD + 16 * hi;
            const unsigned short* b0   = Ps + (n0 + ln) * QPAD + 16 * hi;
            const unsigned short* b1   = Ps + (n0 + 16 + ln) * QPAD + 16 * hi;
            v8f acc0 = VZERO, acc1 = VZERO;
            #pragma unroll
            for (int k0 = 0; k0 < WWW; k0 += 32) {
                FragU a = ld_frag_a(arow + k0);
                acc0 = wmma_bf16(a, ld_frag_b(b0 + k0), acc0);
                acc1 = wmma_bf16(a, ld_frag_b(b1 + k0), acc1);
            }
            float* orow = out
                + (((size_t)b * CC + dbase + m0 + 8 * hi) * HHH + h) * WWW + ln;
            #pragma unroll
            for (int r = 0; r < 8; ++r) {
                orow[(size_t)r * HWSZ + n0]      = acc0[r];
                orow[(size_t)r * HWSZ + n0 + 16] = acc1[r];
            }
        }
        __syncthreads();   // Vs reused next d-block
    }
}

// ---------------------------------------------------------------------------
extern "C" void kernel_launch(void* const* d_in, const int* in_sizes, int n_in,
                              void* d_out, int out_size, void* d_ws, size_t ws_size,
                              hipStream_t stream) {
    const float* x     = (const float*)d_in[0];
    const float* Wq    = (const float*)d_in[1];
    const float* bq    = (const float*)d_in[2];
    const float* Wk    = (const float*)d_in[3];
    const float* bk    = (const float*)d_in[4];
    const float* Wv    = (const float*)d_in[5];
    const float* bv    = (const float*)d_in[6];
    const float* rel_h = (const float*)d_in[7];
    const float* rel_w = (const float*)d_in[8];
    float* out = (float*)d_out;

    unsigned short* wsb = (unsigned short*)d_ws;
    const int nconv = 2 * C8C * CC + CC * CC;          // 327680 elements
    convert_weights_kernel<<<(nconv + 255) / 256, 256, 0, stream>>>(Wq, Wk, Wv, wsb);

    unsigned short* wqb = wsb;
    unsigned short* wkb = wsb + C8C * CC;
    unsigned short* wvb = wsb + 2 * C8C * CC;

    axial_attn_kernel<<<BB * HHH, 256, SMEM_BYTES, stream>>>(
        x, wqb, wkb, wvb, bq, bk, bv, rel_h, rel_w, out);
}
